// LongformerAttention_20813411516821
// MI455X (gfx1250) — matrix operand order repaired and checked by reading, not compiled, not run on hardware
//
#include <hip/hip_runtime.h>

#define H 768
#define NH 12
#define HD 64
#define SEQ 2048
#define BATCH 4
#define MROWS (BATCH * SEQ)  // 8192

typedef unsigned short u16;
typedef unsigned int   u32;
typedef __attribute__((ext_vector_type(16))) __bf16 v16bf;
typedef __attribute__((ext_vector_type(2)))  __bf16 v2bf;
typedef __attribute__((ext_vector_type(8)))  float  v8f;
typedef __attribute__((ext_vector_type(4)))  int    v4i;

union FragB { u32 u[8]; v16bf v; };

#if __has_builtin(__builtin_amdgcn_global_load_async_to_lds_b128) && \
    __has_builtin(__builtin_amdgcn_s_wait_asynccnt)
#define HAVE_ASYNC_LDS 1
typedef __attribute__((address_space(1))) v4i glb_v4i;
typedef __attribute__((address_space(3))) v4i lds_v4i;
__device__ __forceinline__ void async_cp16(const void* g, void* l) {
  // one 16B chunk per lane, global -> LDS, tracked by ASYNCcnt
  __builtin_amdgcn_global_load_async_to_lds_b128(
      (glb_v4i*)(__UINTPTR_TYPE__)g, (lds_v4i*)(__UINTPTR_TYPE__)l, 0, 0);
}
#else
#define HAVE_ASYNC_LDS 0
#endif

#if defined(__gfx1250__) && __has_builtin(__builtin_amdgcn_cvt_pk_bf16_f32)
__device__ __forceinline__ u32 pack2_bf16(float a, float b) {
  union { v2bf v; u32 u; } cv;
  cv.v = __builtin_amdgcn_cvt_pk_bf16_f32(a, b);  // 1 VALU op
  return cv.u;
}
#else
__device__ __forceinline__ u32 pack2_bf16(float a, float b) {
  u32 ua = __float_as_uint(a);
  u32 ub = __float_as_uint(b);
  ua = (ua + 0x7FFFu + ((ua >> 16) & 1u)) >> 16;   // RNE
  ub = (ub + 0x7FFFu + ((ub >> 16) & 1u)) >> 16;
  return (ua & 0xFFFFu) | (ub << 16);
}
#endif

__device__ __forceinline__ u16 bf16_1(float a) {
  u32 ua = __float_as_uint(a);
  return (u16)((ua + 0x7FFFu + ((ua >> 16) & 1u)) >> 16);
}

__device__ __forceinline__ v8f v8f_zero() {
  v8f z;
#pragma unroll
  for (int e = 0; e < 8; ++e) z[e] = 0.0f;
  return z;
}

// ---------------------------------------------------------------------------
// C[M x 768] = (X[M x 768] @ W[768 x 768] + bias) * scale
//   XBF16  : 0 -> X is f32 row-major, 1 -> X is bf16 row-major
//   OUTMODE: 0 -> f32 row-major [M,768], 1 -> bf16 head-split [B, NH, S, HD]
// 256 threads (8 waves), block tile 128x128, each wave 32x64 (2x4 WMMA tiles).
// Software pipeline: global loads of tile t+1 issue before WMMA of tile t.
// ---------------------------------------------------------------------------
template <int XBF16, int OUTMODE>
__global__ __launch_bounds__(256)
void proj_gemm_kernel(const void* __restrict__ Xv,
                      const float* __restrict__ W,
                      const float* __restrict__ bias,
                      void* __restrict__ outv,
                      float scale)
{
  __shared__ u32 As[128][17];  // [m][K-pair]  packed 2 x bf16
  __shared__ u32 Bs[128][17];  // [n][K-pair]  W transposed on store

  const int tid    = threadIdx.x;
  const int lane   = tid & 31;
  const int wid    = tid >> 5;
  const int half   = lane >> 4;
  const int m16    = lane & 15;
  const int wave_m = wid & 3;
  const int wave_n = wid >> 2;

  const int n0 = blockIdx.x * 128;
  const int m0 = blockIdx.y * 128;

  const int pA  = tid & 15;   // K-pair for X staging
  const int mb  = tid >> 4;   // 0..15
  const int nl  = tid & 127;  // column for W staging
  const int pbW = tid >> 7;   // 0..1

  u32 xa[8], wb[8];           // register staging for the next tile

  auto load_stage = [&](int k0) {
    if (XBF16) {
      const u16* X = (const u16*)Xv;
#pragma unroll
      for (int mm = 0; mm < 8; ++mm) {
        int m = mb + mm * 16;
        xa[mm] = *(const u32*)(X + (size_t)(m0 + m) * H + k0 + 2 * pA);
      }
    } else {
      const float* X = (const float*)Xv;
#pragma unroll
      for (int mm = 0; mm < 8; ++mm) {
        int m = mb + mm * 16;
        float2 xv = *(const float2*)(X + (size_t)(m0 + m) * H + k0 + 2 * pA);
        xa[mm] = pack2_bf16(xv.x, xv.y);
      }
    }
#pragma unroll
    for (int pp = 0; pp < 8; ++pp) {
      int p = pbW + pp * 2;
      float w0 = W[(size_t)(k0 + 2 * p) * H + n0 + nl];
      float w1 = W[(size_t)(k0 + 2 * p + 1) * H + n0 + nl];
      wb[pp] = pack2_bf16(w0, w1);
    }
  };
  auto commit_stage = [&]() {
#pragma unroll
    for (int mm = 0; mm < 8; ++mm) As[mb + mm * 16][pA] = xa[mm];
#pragma unroll
    for (int pp = 0; pp < 8; ++pp) Bs[nl][pbW + pp * 2] = wb[pp];
  };

  v8f acc[2][4];
#pragma unroll
  for (int r = 0; r < 2; ++r)
#pragma unroll
    for (int c = 0; c < 4; ++c) acc[r][c] = v8f_zero();

  load_stage(0);
  for (int k0 = 0; k0 < H; k0 += 32) {
    commit_stage();
    __syncthreads();
    if (k0 + 32 < H) load_stage(k0 + 32);  // overlap with WMMA below

    FragB a[2];
#pragma unroll
    for (int r = 0; r < 2; ++r)
#pragma unroll
      for (int j = 0; j < 8; ++j) {
        int pa = j + half * 8;  // A linear pairs
        a[r].u[j] = As[wave_m * 32 + r * 16 + m16][pa];
      }
#pragma unroll
    for (int c = 0; c < 4; ++c) {
      FragB b;
#pragma unroll
      for (int j = 0; j < 8; ++j)
        b.u[j] = Bs[wave_n * 64 + c * 16 + m16][j + half * 8];  // B linear pairs
#pragma unroll
      for (int r = 0; r < 2; ++r)
        acc[r][c] = __builtin_amdgcn_wmma_f32_16x16x32_bf16(
            false, a[r].v, false, b.v, (short)0, acc[r][c], false, false);
    }
    __syncthreads();
  }

#pragma unroll
  for (int c = 0; c < 4; ++c) {
    int gn = n0 + wave_n * 64 + c * 16 + m16;
    float bv = bias[gn];
#pragma unroll
    for (int r = 0; r < 2; ++r) {
#pragma unroll
      for (int vr = 0; vr < 8; ++vr) {
        int gm = m0 + wave_m * 32 + r * 16 + vr + half * 8;
        float val = (acc[r][c][vr] + bv) * scale;
        if (OUTMODE == 0) {
          ((float*)outv)[(size_t)gm * H + gn] = val;
        } else {
          int bb = gm >> 11, s = gm & (SEQ - 1);
          int h = gn >> 6, d = gn & (HD - 1);
          ((u16*)outv)[(((size_t)(bb * NH + h)) * SEQ + s) * HD + d] = bf16_1(val);
        }
      }
    }
  }
}

// ---------------------------------------------------------------------------
// Banded attention on bf16 q/k/v [B, NH, S, HD]; q pre-scaled by 1/8.
// One block = one (b,h) x 64 queries, 128-key window [i0-32, i0+95].
// 128 threads = 4 waves, each wave owns a 16-query strip.
// ctx written bf16 row-major [B, S, NH*HD].
// ---------------------------------------------------------------------------
__global__ __launch_bounds__(128)
void band_attn_kernel(const u16* __restrict__ q,
                      const u16* __restrict__ k,
                      const u16* __restrict__ v,
                      u16* __restrict__ ctx)
{
  __shared__ u32 Qs[64][17];   // [query][d-pair]
  __shared__ u32 Ks[128][17];  // [key][d-pair]
  __shared__ u32 Vs[64][65];   // [d][key-pair] (transposed)
  __shared__ u16 Ps[64][132];  // probs bf16 [query][key]

  const int tid  = threadIdx.x;
  const int lane = tid & 31;
  const int wid  = tid >> 5;
  const int half = lane >> 4;
  const int m16  = lane & 15;

  const int i0 = blockIdx.x * 64;
  const int bh = blockIdx.y;     // b*NH + h
  const int j0 = i0 - 32;

  const u16* qp = q + (size_t)bh * SEQ * HD;
  const u16* kp = k + (size_t)bh * SEQ * HD;
  const u16* vp = v + (size_t)bh * SEQ * HD;

#if HAVE_ASYNC_LDS
  // Q: 64 rows x 64B, 16B chunks, async global->LDS (no VGPR round trip)
#pragma unroll
  for (int it = 0; it < 2; ++it) {
    int idx = tid + it * 128;
    int row = idx >> 2, c4 = idx & 3;
    async_cp16(qp + (size_t)(i0 + row) * HD + c4 * 8, &Qs[row][c4 * 4]);
  }
  // K: 128 rows x 64B with OOB zero-fill
#pragma unroll
  for (int it = 0; it < 4; ++it) {
    int idx = tid + it * 128;
    int jl = idx >> 2, c4 = idx & 3;
    int j = j0 + jl;
    if (j >= 0 && j < SEQ) {
      async_cp16(kp + (size_t)j * HD + c4 * 8, &Ks[jl][c4 * 4]);
    } else {
      u32* d = &Ks[jl][c4 * 4];
      d[0] = 0; d[1] = 0; d[2] = 0; d[3] = 0;
    }
  }
#else
  {
    int p = tid & 15, rb = tid >> 4;
#pragma unroll
    for (int rr = 0; rr < 8; ++rr) {
      int r = rb + rr * 8;
      Qs[r][p] = *(const u32*)(qp + (size_t)(i0 + r) * HD + 2 * p);
    }
#pragma unroll
    for (int jj = 0; jj < 16; ++jj) {
      int jl = (tid >> 4) + jj * 8;
      int j = j0 + jl;
      Ks[jl][p] = (j >= 0 && j < SEQ)
                      ? *(const u32*)(kp + (size_t)j * HD + 2 * p) : 0u;
    }
  }
#endif
  // V transposed: Vs[d][jp] = {v[j0+2jp][d], v[j0+2jp+1][d]}
  {
    int d = tid & 63, jpb = tid >> 6;
#pragma unroll
    for (int it = 0; it < 32; ++it) {
      int jp = jpb + it * 2;
      int ja = j0 + 2 * jp, jb2 = ja + 1;
      u32 lo = (ja  >= 0 && ja  < SEQ) ? (u32)vp[(size_t)ja  * HD + d] : 0u;
      u32 hi = (jb2 >= 0 && jb2 < SEQ) ? (u32)vp[(size_t)jb2 * HD + d] : 0u;
      Vs[d][jp] = lo | (hi << 16);
    }
  }
#if HAVE_ASYNC_LDS
  __builtin_amdgcn_s_wait_asynccnt(0);
#endif
  __syncthreads();

  // scores S = Q*K^T  (16 x 128 per wave)
  v8f sc[8];
#pragma unroll
  for (int c = 0; c < 8; ++c) sc[c] = v8f_zero();

#pragma unroll
  for (int dk = 0; dk < HD; dk += 32) {
    FragB a;
#pragma unroll
    for (int j = 0; j < 8; ++j) {
      int pa = (dk >> 1) + j + half * 8;
      a.u[j] = Qs[wid * 16 + m16][pa];
    }
#pragma unroll
    for (int c = 0; c < 8; ++c) {
      FragB b;
#pragma unroll
      for (int j = 0; j < 8; ++j)
        b.u[j] = Ks[c * 16 + m16][(dk >> 1) + j + half * 8];
      sc[c] = __builtin_amdgcn_wmma_f32_16x16x32_bf16(
          false, a.v, false, b.v, (short)0, sc[c], false, false);
    }
  }

  // band mask + softmax (row = vr + 8*half across the 16-lane half)
  float inv_rs[8];
#pragma unroll
  for (int vr = 0; vr < 8; ++vr) {
    int irow = i0 + wid * 16 + vr + half * 8;
    float mx = -3.0e38f;
#pragma unroll
    for (int c = 0; c < 8; ++c) {
      int j = j0 + c * 16 + m16;
      int dd = irow - j;
      bool ok = (j >= 0) && (j < SEQ) && (dd <= 32) && (dd >= -32);
      float sv = ok ? sc[c][vr] : -1.0e30f;
      sc[c][vr] = sv;
      mx = fmaxf(mx, sv);
    }
#pragma unroll
    for (int m = 1; m <= 8; m <<= 1) mx = fmaxf(mx, __shfl_xor(mx, m, 32));
    float rs = 0.0f;
#pragma unroll
    for (int c = 0; c < 8; ++c) {
      float pv = __builtin_amdgcn_exp2f((sc[c][vr] - mx) * 1.44269504088896340736f);
      sc[c][vr] = pv;
      rs += pv;
    }
#pragma unroll
    for (int m = 1; m <= 8; m <<= 1) rs += __shfl_xor(rs, m, 32);
    inv_rs[vr] = 1.0f / rs;
#pragma unroll
    for (int c = 0; c < 8; ++c)
      Ps[wid * 16 + vr + half * 8][c * 16 + m16] = bf16_1(sc[c][vr]);
  }
  __syncthreads();

  // ctx = P * V   (16 x 64 per wave)
  v8f acc2[4];
#pragma unroll
  for (int c2 = 0; c2 < 4; ++c2) acc2[c2] = v8f_zero();

#pragma unroll
  for (int jk = 0; jk < 128; jk += 32) {
    FragB a;
#pragma unroll
    for (int j = 0; j < 8; ++j) {
      int pa = (jk >> 1) + j + half * 8;
      a.u[j] = *(const u32*)&Ps[wid * 16 + m16][2 * pa];
    }
#pragma unroll
    for (int c2 = 0; c2 < 4; ++c2) {
      FragB b;
#pragma unroll
      for (int j = 0; j < 8; ++j)
        b.u[j] = Vs[c2 * 16 + m16][(jk >> 1) + j + half * 8];
      acc2[c2] = __builtin_amdgcn_wmma_f32_16x16x32_bf16(
          false, a.v, false, b.v, (short)0, acc2[c2], false, false);
    }
  }

  const int bidx = bh / NH, h = bh % NH;
#pragma unroll
  for (int c2 = 0; c2 < 4; ++c2) {
    int d = c2 * 16 + m16;
#pragma unroll
    for (int vr = 0; vr < 8; ++vr) {
      int srow = i0 + wid * 16 + vr + half * 8;
      float val = acc2[c2][vr] * inv_rs[vr];
      ctx[((size_t)(bidx * SEQ + srow)) * H + h * HD + d] = bf16_1(val);
    }
  }
}

// ---------------------------------------------------------------------------
extern "C" void kernel_launch(void* const* d_in, const int* in_sizes, int n_in,
                              void* d_out, int out_size, void* d_ws, size_t ws_size,
                              hipStream_t stream)
{
  const float* x  = (const float*)d_in[0];
  const float* Wq = (const float*)d_in[1];
  const float* bq = (const float*)d_in[2];
  const float* Wk = (const float*)d_in[3];
  const float* bk = (const float*)d_in[4];
  const float* Wv = (const float*)d_in[5];
  const float* bv = (const float*)d_in[6];
  const float* Wo = (const float*)d_in[7];
  const float* bo = (const float*)d_in[8];
  float* out = (float*)d_out;

  const size_t elems = (size_t)MROWS * H;
  u16* qbuf = (u16*)d_ws;        // bf16 intermediates: 4 * 12.6 MB = 50.3 MB
  u16* kbuf = qbuf + elems;
  u16* vbuf = kbuf + elems;
  u16* cbuf = vbuf + elems;      // ctx bf16 row-major [M, H]

  dim3 pg(H / 128, MROWS / 128);  // 6 x 64
  // 1/sqrt(HD)=0.125 folded into the Q projection
  proj_gemm_kernel<0, 1><<<pg, 256, 0, stream>>>(x, Wq, bq, qbuf, 0.125f);
  proj_gemm_kernel<0, 1><<<pg, 256, 0, stream>>>(x, Wk, bk, kbuf, 1.0f);
  proj_gemm_kernel<0, 1><<<pg, 256, 0, stream>>>(x, Wv, bv, vbuf, 1.0f);

  dim3 ag(SEQ / 64, BATCH * NH);  // 32 x 48
  band_attn_kernel<<<ag, 128, 0, stream>>>(qbuf, kbuf, vbuf, cbuf);

  proj_gemm_kernel<1, 0><<<pg, 256, 0, stream>>>(cbuf, Wo, bo, out, 1.0f);
}
